// RingDilatedAttentionTritonIntegrated_68959994905213
// MI455X (gfx1250) — compile-verified, matrix operand order checked by reading, not run on hardware
//
#include <hip/hip_runtime.h>

typedef __attribute__((ext_vector_type(16))) __bf16 v16bf;
typedef __attribute__((ext_vector_type(8)))  __bf16 v8bf;
typedef __attribute__((ext_vector_type(8)))  float  v8f;

#define LDSS 72          // bf16 elements per padded LDS row (64 + 8) -> 144B, staggers banks
#define NWAVE 16
#define RPW   32         // query rows per wave

// scale (1/sqrt(64)) * log2(e): softmax runs in exp2 domain
#define QSCALE 0.18033688011112042f

// 16-lane reductions: single-instruction VOP2+DPP steps (max(dpp(x), x)).
// xor1/2/4 via DPP8 permutes within 8 lanes; xor8 via DPP16 row_ror:8
// (rotate by 8 within a row of 16 == xor 8).
static __device__ __forceinline__ float rmax16(float x) {
  float t;
  asm("v_max_num_f32_dpp %0, %1, %1 dpp8:[1,0,3,2,5,4,7,6]" : "=v"(t) : "v"(x));
  asm("v_max_num_f32_dpp %0, %1, %1 dpp8:[2,3,0,1,6,7,4,5]" : "=v"(x) : "v"(t));
  asm("v_max_num_f32_dpp %0, %1, %1 dpp8:[4,5,6,7,0,1,2,3]" : "=v"(t) : "v"(x));
  asm("v_max_num_f32_dpp %0, %1, %1 row_ror:8 row_mask:0xf bank_mask:0xf bound_ctrl:1"
      : "=v"(x) : "v"(t));
  return x;
}
static __device__ __forceinline__ float rsum16(float x) {
  float t;
  asm("v_add_f32_dpp %0, %1, %1 dpp8:[1,0,3,2,5,4,7,6]" : "=v"(t) : "v"(x));
  asm("v_add_f32_dpp %0, %1, %1 dpp8:[2,3,0,1,6,7,4,5]" : "=v"(x) : "v"(t));
  asm("v_add_f32_dpp %0, %1, %1 dpp8:[4,5,6,7,0,1,2,3]" : "=v"(t) : "v"(x));
  asm("v_add_f32_dpp %0, %1, %1 row_ror:8 row_mask:0xf bank_mask:0xf bound_ctrl:1"
      : "=v"(x) : "v"(t));
  return x;
}

static __device__ __forceinline__ v16bf ld_frag(const __bf16* p) {
  // A/B fragment: two contiguous runs of 8 bf16 (16B each), 16 elements apart
  v8bf a = *(const v8bf*)(p);
  v8bf b = *(const v8bf*)(p + 16);
  return __builtin_shufflevector(a, b, 0,1,2,3,4,5,6,7,8,9,10,11,12,13,14,15);
}

struct TileRegs { float4 k0, k1, v0, v1; };

__global__ void __launch_bounds__(512) zero_f32(float* p, int n) {
  int i = blockIdx.x * 512 + threadIdx.x;
  if (i < n) p[i] = 0.0f;
}

__global__ void __launch_bounds__(512)
dilated_flash_attn(const float* __restrict__ q, const float* __restrict__ k,
                   const float* __restrict__ v, float* __restrict__ out) {
  __shared__ __bf16 Ksh[64 * LDSS];             // key tile, row-major (key x dim)
  __shared__ __bf16 VshT[64 * LDSS];            // value tile, transposed (dim x key)
  __shared__ __bf16 Scr[NWAVE * 16 * LDSS];     // per-wave 16-row scratch (Q stage / P)

  // ---- map block -> (config, segment, head) : 64 + 32 + 16 = 112 blocks ----
  const int bid = blockIdx.x;
  int seg_len, r, s, h;
  if (bid < 64)      { seg_len = 2048; r = 4;  s = bid >> 4;        h = bid & 15;        }
  else if (bid < 96) { seg_len = 4096; r = 8;  s = (bid - 64) >> 4; h = (bid - 64) & 15; }
  else               { seg_len = 8192; r = 16; s = 0;               h = bid - 96;        }

  const int rowStride = r * 1024;                           // floats between gathered rows (H*D=1024)
  const int origin    = (s * seg_len + (s % r)) * 1024 + h * 64;

  const int tid  = threadIdx.x;
  const int wave = tid >> 5;
  const int lane = tid & 31;
  const int lh   = lane >> 4;      // half-wave
  const int ll   = lane & 15;

  __bf16* scr = &Scr[wave * 16 * LDSS];

  // ---- stage Q (x scale*log2e) and build A-fragments: 2 M-tiles x 2 K-chunks ----
  v16bf qa[2][2];
#pragma unroll
  for (int mt = 0; mt < 2; ++mt) {
    const int qIdx  = wave * RPW + mt * 16 + ll;            // global query row 0..511
    const int dbase = lh * 32;                              // each lane: 1 row, 32 dims
    const float4* qp = (const float4*)(q + origin + qIdx * rowStride + dbase);
#pragma unroll
    for (int c = 0; c < 4; ++c) {
      float4 a = qp[2 * c], b = qp[2 * c + 1];
      v8bf pk;
      pk[0] = (__bf16)(a.x * QSCALE); pk[1] = (__bf16)(a.y * QSCALE);
      pk[2] = (__bf16)(a.z * QSCALE); pk[3] = (__bf16)(a.w * QSCALE);
      pk[4] = (__bf16)(b.x * QSCALE); pk[5] = (__bf16)(b.y * QSCALE);
      pk[6] = (__bf16)(b.z * QSCALE); pk[7] = (__bf16)(b.w * QSCALE);
      *(v8bf*)&scr[ll * LDSS + dbase + c * 8] = pk;
    }
    __builtin_amdgcn_wave_barrier();                        // same-wave LDS is in-order
#pragma unroll
    for (int kc = 0; kc < 2; ++kc)
      qa[mt][kc] = ld_frag(&scr[ll * LDSS + kc * 32 + lh * 8]);
    __builtin_amdgcn_wave_barrier();
  }

  // ---- accumulators / online-softmax state (C-layout: VGPR vv <-> rows vv, 8+vv) ----
  v8f   oacc[2][4];
  float runM[2][8], runS[2][8];
#pragma unroll
  for (int mt = 0; mt < 2; ++mt)
#pragma unroll
    for (int vv = 0; vv < 8; ++vv) {
      runM[mt][vv] = -3.0e38f; runS[mt][vv] = 0.0f;
#pragma unroll
      for (int dt = 0; dt < 4; ++dt) oacc[mt][dt][vv] = 0.0f;
    }

  // ---- software-pipelined flash loop over 8 key tiles of 64 ----
  // Register double-buffer: global loads for tile kt+1 are issued before the
  // compute of tile kt, so HBM/L2 latency hides under the WMMA phase.
  const int lrow = tid >> 3, ldseg = (tid & 7) * 8;   // 512 thr: 64 rows x 8 dims
  const float* kbase = k + origin + lrow * rowStride + ldseg;
  const float* vbase = v + origin + lrow * rowStride + ldseg;

  TileRegs tr;
  tr.k0 = ((const float4*)kbase)[0]; tr.k1 = ((const float4*)kbase)[1];
  tr.v0 = ((const float4*)vbase)[0]; tr.v1 = ((const float4*)vbase)[1];

  for (int kt = 0; kt < 8; ++kt) {
    __syncthreads();                                  // all waves done with prev tile
    {   // convert in-flight registers -> bf16 LDS tile
      v8bf kk;
      kk[0] = (__bf16)tr.k0.x; kk[1] = (__bf16)tr.k0.y;
      kk[2] = (__bf16)tr.k0.z; kk[3] = (__bf16)tr.k0.w;
      kk[4] = (__bf16)tr.k1.x; kk[5] = (__bf16)tr.k1.y;
      kk[6] = (__bf16)tr.k1.z; kk[7] = (__bf16)tr.k1.w;
      *(v8bf*)&Ksh[lrow * LDSS + ldseg] = kk;
      VshT[(ldseg + 0) * LDSS + lrow] = (__bf16)tr.v0.x;
      VshT[(ldseg + 1) * LDSS + lrow] = (__bf16)tr.v0.y;
      VshT[(ldseg + 2) * LDSS + lrow] = (__bf16)tr.v0.z;
      VshT[(ldseg + 3) * LDSS + lrow] = (__bf16)tr.v0.w;
      VshT[(ldseg + 4) * LDSS + lrow] = (__bf16)tr.v1.x;
      VshT[(ldseg + 5) * LDSS + lrow] = (__bf16)tr.v1.y;
      VshT[(ldseg + 6) * LDSS + lrow] = (__bf16)tr.v1.z;
      VshT[(ldseg + 7) * LDSS + lrow] = (__bf16)tr.v1.w;
    }
    if (kt + 1 < 8) {                                 // issue next tile's loads now
      const float* kp = kbase + (kt + 1) * 64 * rowStride;
      const float* vp = vbase + (kt + 1) * 64 * rowStride;
      tr.k0 = ((const float4*)kp)[0]; tr.k1 = ((const float4*)kp)[1];
      tr.v0 = ((const float4*)vp)[0]; tr.v1 = ((const float4*)vp)[1];
    }
    __syncthreads();                                  // tile kt visible in LDS

#pragma unroll
    for (int mt = 0; mt < 2; ++mt) {
      // S = Q * K^T  (4 N-tiles x 2 K-chunks); per-use fragment loads — the
      // scheduler issues ds_load_b128s ahead with staggered dscnt waits.
      v8f sacc[4];
#pragma unroll
      for (int nt = 0; nt < 4; ++nt)
#pragma unroll
        for (int vv = 0; vv < 8; ++vv) sacc[nt][vv] = 0.0f;
#pragma unroll
      for (int kc = 0; kc < 2; ++kc)
#pragma unroll
        for (int nt = 0; nt < 4; ++nt) {
          v16bf kb = ld_frag(&Ksh[(nt * 16 + ll) * LDSS + kc * 32 + lh * 8]);
          sacc[nt] = __builtin_amdgcn_wmma_f32_16x16x32_bf16(
              false, qa[mt][kc], false, kb, (short)0, sacc[nt], false, false);
        }

      // online softmax (exp2 domain); write P (bf16) to per-wave scratch
#pragma unroll
      for (int vv = 0; vv < 8; ++vv) {
        float mx = rmax16(fmaxf(fmaxf(sacc[0][vv], sacc[1][vv]),
                                fmaxf(sacc[2][vv], sacc[3][vv])));
        const float om = runM[mt][vv];
        const float nm = fmaxf(om, mx);
        const float cf = __builtin_amdgcn_exp2f(om - nm);
        runM[mt][vv] = nm;
        float ls = 0.0f;
#pragma unroll
        for (int nt = 0; nt < 4; ++nt) {
          float p = __builtin_amdgcn_exp2f(sacc[nt][vv] - nm);
          sacc[nt][vv] = p;
          ls += p;
        }
        ls = rsum16(ls);
        runS[mt][vv] = runS[mt][vv] * cf + ls;
#pragma unroll
        for (int dt = 0; dt < 4; ++dt) oacc[mt][dt][vv] *= cf;
        const int prow = lh * 8 + vv;
#pragma unroll
        for (int nt = 0; nt < 4; ++nt)
          scr[prow * LDSS + nt * 16 + ll] = (__bf16)sacc[nt][vv];
      }
      __builtin_amdgcn_wave_barrier();

      // O += P * V  (2 K-chunks x 4 D-tiles), per-use fragment loads
#pragma unroll
      for (int kc = 0; kc < 2; ++kc) {
        v16bf pa = ld_frag(&scr[ll * LDSS + kc * 32 + lh * 8]);
#pragma unroll
        for (int dt = 0; dt < 4; ++dt) {
          v16bf vb = ld_frag(&VshT[(dt * 16 + ll) * LDSS + kc * 32 + lh * 8]);
          oacc[mt][dt] = __builtin_amdgcn_wmma_f32_16x16x32_bf16(
              false, pa, false, vb, (short)0, oacc[mt][dt], false, false);
        }
      }
      __builtin_amdgcn_wave_barrier();
    }
  }

  // ---- epilogue: O / rowsum * (1/3), scatter-accumulate (configs overlap) ----
  const float third = 1.0f / 3.0f;
#pragma unroll
  for (int mt = 0; mt < 2; ++mt)
#pragma unroll
    for (int vv = 0; vv < 8; ++vv) {
      const int qIdx = wave * RPW + mt * 16 + lh * 8 + vv;
      float* op = out + origin + qIdx * rowStride;
      const float invs = third * __builtin_amdgcn_rcpf(runS[mt][vv]);
#pragma unroll
      for (int dt = 0; dt < 4; ++dt)
        unsafeAtomicAdd(op + dt * 16 + ll, oacc[mt][dt][vv] * invs);
    }
}

extern "C" void kernel_launch(void* const* d_in, const int* in_sizes, int n_in,
                              void* d_out, int out_size, void* d_ws, size_t ws_size,
                              hipStream_t stream) {
  const float* q = (const float*)d_in[0];
  const float* k = (const float*)d_in[1];
  const float* v = (const float*)d_in[2];
  float* out = (float*)d_out;
  zero_f32<<<(out_size + 511) / 512, 512, 0, stream>>>(out, out_size);
  dilated_flash_attn<<<112, 512, 0, stream>>>(q, k, v, out);
}
